// CellGraphTransformer_5119601017145
// MI455X (gfx1250) — matrix operand order repaired and checked by reading, not compiled, not run on hardware
//
#include <hip/hip_runtime.h>
#include <hip/hip_bf16.h>
#include <math.h>

// ---------- vector types ----------
typedef __attribute__((ext_vector_type(16))) __bf16          v16bf;
typedef __attribute__((ext_vector_type(8)))  float           v8f;
typedef __attribute__((ext_vector_type(16))) unsigned short  u16x16;
typedef __attribute__((ext_vector_type(8)))  unsigned short  u16x8;
typedef __attribute__((ext_vector_type(4)))  unsigned short  u16x4;
typedef __attribute__((ext_vector_type(4)))  float           f32x4;
typedef __attribute__((ext_vector_type(4)))  unsigned int    u32x4;
typedef __attribute__((ext_vector_type(8)))  int             i32x8;
typedef __attribute__((ext_vector_type(4)))  int             i32x4;

#ifndef USE_TDM
#if __has_builtin(__builtin_amdgcn_tensor_load_to_lds)
#define USE_TDM 1
#else
#define USE_TDM 0
#endif
#endif

// ---------- helpers ----------
__device__ __forceinline__ unsigned short f2bf(float f) {
  unsigned int u = __builtin_bit_cast(unsigned int, f);
  u += 0x7FFFu + ((u >> 16) & 1u);            // round-to-nearest-even
  return (unsigned short)(u >> 16);
}

__device__ __forceinline__ u16x4 f4tobf(f32x4 f) {
  u16x4 r;
  r[0] = f2bf(f[0]); r[1] = f2bf(f[1]); r[2] = f2bf(f[2]); r[3] = f2bf(f[3]);
  return r;
}

__device__ __forceinline__ u16x16 cat8(u16x8 lo, u16x8 hi) {
  u16x16 r;
#pragma unroll
  for (int i = 0; i < 8; ++i) { r[i] = lo[i]; r[i + 8] = hi[i]; }
  return r;
}

__device__ __forceinline__ v8f wmma_bf16(u16x16 a, u16x16 b, v8f c) {
  return __builtin_amdgcn_wmma_f32_16x16x32_bf16(
      false, __builtin_bit_cast(v16bf, a),
      false, __builtin_bit_cast(v16bf, b),
      (short)0, c, false, false);
}

__device__ __forceinline__ void wait_tensorcnt0() {
#if __has_builtin(__builtin_amdgcn_s_wait_tensorcnt)
  __builtin_amdgcn_s_wait_tensorcnt(0);
#else
  asm volatile("s_wait_tensorcnt 0x0" ::: "memory");
#endif
}
__device__ __forceinline__ void wait_tensorcnt2() {
#if __has_builtin(__builtin_amdgcn_s_wait_tensorcnt)
  __builtin_amdgcn_s_wait_tensorcnt(2);
#else
  asm volatile("s_wait_tensorcnt 0x2" ::: "memory");
#endif
}

#if USE_TDM
// Issue one TDM 2D tile load: 128 rows x 64 bf16, row pitch K elements,
// LDS rows padded by TDM (pad every 32 dwords by 4 dwords) -> 144B stride.
__device__ __forceinline__ void tdm_load_tile(const unsigned short* gbase, int K,
                                              int rows_total, unsigned int lds_off) {
  unsigned long long ga = (unsigned long long)(const void*)gbase;
  u32x4 g0;
  g0[0] = 1u;                                       // count=1 valid descriptor
  g0[1] = lds_off;                                  // LDS byte address
  g0[2] = (unsigned int)(ga & 0xFFFFFFFFull);       // global addr [31:0]
  g0[3] = (unsigned int)((ga >> 32) & 0x1FFFFFFull) // global addr [56:32]
          | (2u << 30);                             // type = 2 ("image")
  i32x8 g1;
  g1[0] = (int)((1u << 16)      // data_size = 1 -> 2 bytes (bf16)
              | (1u << 20)      // pad_enable
              | (4u << 22)      // pad_interval: 32 dwords (=128B tile row)
              | (3u << 25));    // pad_amount: 4 dwords (=16B pad)
  g1[1] = (int)(((unsigned)K & 0xFFFFu) << 16);                       // tensor_dim0 lo
  g1[2] = (int)((((unsigned)K >> 16) & 0xFFFFu)                        // tensor_dim0 hi
              | (((unsigned)rows_total & 0xFFFFu) << 16));             // tensor_dim1 lo
  g1[3] = (int)(((((unsigned)rows_total) >> 16) & 0xFFFFu)             // tensor_dim1 hi
              | (64u << 16));                                          // tile_dim0 = 64
  g1[4] = 128;                                                         // tile_dim1 = 128
  g1[5] = K;                                                           // dim0_stride lo
  g1[6] = 0;
  g1[7] = 0;
  i32x4 gz = {0, 0, 0, 0};
#if defined(__clang_major__) && (__clang_major__ >= 23)
  i32x8 gz8 = {0, 0, 0, 0, 0, 0, 0, 0};
  __builtin_amdgcn_tensor_load_to_lds(g0, g1, gz, gz, gz8, 0);
#else
  __builtin_amdgcn_tensor_load_to_lds(g0, g1, gz, gz, 0);
#endif
}
#endif

// =====================================================================
// bf16 WMMA GEMM: C[M,N] = act(A[M,K] @ Bt[N,K]^T + bias [+ R])
// A bf16 row-major, Bt bf16 pre-transposed [N,K].
// 256 threads (8 waves), block tile 128x128, k-step 64, LDS double-buffered;
// tiles staged by the Tensor Data Mover (wave 0), prefetching tile i+1
// while tile i is consumed by 16 WMMAs/wave.
// =====================================================================
#define BM  128
#define BN  128
#define BK2 64
#define LST 72    // LDS row stride in elements (144B: 16B-aligned, bank-spread)

__global__ __launch_bounds__(256)
void gemm_bf16t_wmma(const unsigned short* __restrict__ A,
                     const unsigned short* __restrict__ Bt,
                     const float* __restrict__ bias, const float* __restrict__ R,
                     float* __restrict__ C, unsigned short* __restrict__ Cbf,
                     int M, int N, int K, int gelu)
{
  __shared__ __align__(16) unsigned short sA[2][BM * LST];
  __shared__ __align__(16) unsigned short sB[2][BN * LST];

  const int t    = threadIdx.x;
  const int lane = t & 31;
  const int wid  = t >> 5;
  const int wm   = wid >> 1;          // 0..3
  const int wn   = wid & 1;           // 0..1
  const int bm   = blockIdx.y * BM;
  const int bn   = blockIdx.x * BN;
  const int l15  = lane & 15;
  const int hi   = (lane >> 4) & 1;
  const int kbA  = hi ? 8 : 0;
  const int kbB  = hi ? 16 : 0;

  v8f zf = {};
  v8f acc[2][4];
#pragma unroll
  for (int i = 0; i < 2; ++i)
#pragma unroll
    for (int j = 0; j < 4; ++j) acc[i][j] = zf;

  const int nk = K / BK2;

#if USE_TDM
  if (wid == 0) {
    tdm_load_tile(A  + (size_t)bm * K, K, M, (unsigned int)(size_t)(void*)&sA[0][0]);
    tdm_load_tile(Bt + (size_t)bn * K, K, N, (unsigned int)(size_t)(void*)&sB[0][0]);
  }
#endif

  for (int i = 0; i < nk; ++i) {
    const int cur = i & 1;
    __syncthreads();                       // prev compute done on both buffers
#if USE_TDM
    if (wid == 0) {
      if (i + 1 < nk) {
        const int k0n = (i + 1) * BK2;
        tdm_load_tile(A  + (size_t)bm * K + k0n, K, M,
                      (unsigned int)(size_t)(void*)&sA[(i + 1) & 1][0]);
        tdm_load_tile(Bt + (size_t)bn * K + k0n, K, N,
                      (unsigned int)(size_t)(void*)&sB[(i + 1) & 1][0]);
        wait_tensorcnt2();                 // tile i complete (in-order TENSORcnt)
      } else {
        wait_tensorcnt0();
      }
    }
#else
    {
      const int k0 = i * BK2;
#pragma unroll
      for (int j = 0; j < 4; ++j) {
        int idx = t + j * 256;             // 0..1023
        int row = idx >> 3;                // 0..127
        int g8  = (idx & 7) * 8;           // 0..56
        *(u16x8*)&sA[cur][row * LST + g8] =
            *(const u16x8*)(A + (size_t)(bm + row) * K + k0 + g8);
        *(u16x8*)&sB[cur][row * LST + g8] =
            *(const u16x8*)(Bt + (size_t)(bn + row) * K + k0 + g8);
      }
    }
#endif
    __syncthreads();                       // tile i visible to all waves

#pragma unroll
    for (int ks = 0; ks < BK2; ks += 32) {
      u16x16 aF[2], bF[4];
#pragma unroll
      for (int mt = 0; mt < 2; ++mt) {
        int m = wm * 32 + mt * 16 + l15;
        const unsigned short* p = &sA[cur][m * LST + ks + kbA];
        aF[mt] = cat8(*(const u16x8*)p, *(const u16x8*)(p + 16));
      }
#pragma unroll
      for (int nt = 0; nt < 4; ++nt) {
        int n = wn * 64 + nt * 16 + l15;
        const unsigned short* p = &sB[cur][n * LST + ks + kbA];
        bF[nt] = cat8(*(const u16x8*)p, *(const u16x8*)(p + 16));
      }
#pragma unroll
      for (int mt = 0; mt < 2; ++mt)
#pragma unroll
        for (int nt = 0; nt < 4; ++nt)
          acc[mt][nt] = wmma_bf16(aF[mt], bF[nt], acc[mt][nt]);
    }
  }

  // ---- epilogue: bias [+residual] [+GELU]; fp32 and/or bf16 outputs ----
#pragma unroll
  for (int mt = 0; mt < 2; ++mt) {
#pragma unroll
    for (int nt = 0; nt < 4; ++nt) {
      int gn = bn + wn * 64 + nt * 16 + l15;
      float bv = bias[gn];
#pragma unroll
      for (int i = 0; i < 8; ++i) {
        int gm = bm + wm * 32 + mt * 16 + i + hi * 8;
        float v = acc[mt][nt][i] + bv;
        if (R)    v += R[(size_t)gm * N + gn];
        if (gelu) v = 0.5f * v * (1.0f + erff(v * 0.70710678118654752f));
        if (C)    C[(size_t)gm * N + gn] = v;
        if (Cbf)  Cbf[(size_t)gm * N + gn] = f2bf(v);
      }
    }
  }
}

// =====================================================================
// Flash attention (head_dim 64), bf16 Q/K/V in, bf16 out.
// 128 threads = 4 waves x 16 queries; 32 keys/chunk staged in LDS.
// grid = (S/64, H, B)
// =====================================================================
#define KC 32

__global__ __launch_bounds__(128)
void flash_attn_wmma(const unsigned short* __restrict__ Q,
                     const unsigned short* __restrict__ Kg,
                     const unsigned short* __restrict__ V,
                     unsigned short* __restrict__ O,
                     int S, int D)
{
  __shared__ __align__(16) unsigned short kbuf[32 * 72];     // [key][d]
  __shared__ __align__(16) unsigned short vbT [64 * 40];     // [d][key]
  __shared__ __align__(16) unsigned short pbuf[4 * 16 * 40]; // per-wave P

  const int t    = threadIdx.x;
  const int lane = t & 31;
  const int w    = t >> 5;
  const int l15  = lane & 15;
  const int hi   = (lane >> 4) & 1;
  const int b    = blockIdx.z;
  const int h    = blockIdx.y;
  const int q0   = blockIdx.x * 64 + w * 16;
  const int kb0  = hi ? 8 : 0;
  const int kb2  = hi ? 16 : 0;

  const size_t headoff = (size_t)b * S * D + (size_t)h * 64;

  // ---- Q fragments (bf16, direct b128 loads) ----
  u16x16 qA[2];
  {
    const unsigned short* qp = Q + headoff + (size_t)(q0 + l15) * D;
#pragma unroll
    for (int f = 0; f < 2; ++f)
      qA[f] = cat8(*(const u16x8*)(qp + f * 32 + kb0),
                   *(const u16x8*)(qp + f * 32 + kb0 + 16));
  }

  v8f zf = {};
  v8f accO[4];
#pragma unroll
  for (int f = 0; f < 4; ++f) accO[f] = zf;
  float runM[8], runL[8];
#pragma unroll
  for (int i = 0; i < 8; ++i) { runM[i] = -3.0e38f; runL[i] = 0.0f; }

  const int nch = S / KC;
  for (int kc = 0; kc < nch; ++kc) {
    __syncthreads();
    // ---- stage K chunk: kbuf[key][d] (pure bf16 copy) ----
#pragma unroll
    for (int j = 0; j < 2; ++j) {
      int idx = t + j * 128;               // 0..255
      int key = idx >> 3;                  // 0..31
      int g8  = (idx & 7) * 8;             // 0..56
      *(u16x8*)&kbuf[key * 72 + g8] =
          *(const u16x8*)(Kg + headoff + (size_t)(kc * KC + key) * D + g8);
    }
    // ---- stage V chunk transposed: vbT[d][key] ----
#pragma unroll
    for (int j = 0; j < 4; ++j) {
      int idx = t + j * 128;               // 0..511
      int key = idx >> 4;                  // 0..31
      int d4  = (idx & 15) * 4;            // 0..60
      u16x4 f = *(const u16x4*)(V + headoff + (size_t)(kc * KC + key) * D + d4);
#pragma unroll
      for (int e = 0; e < 4; ++e) vbT[(d4 + e) * 40 + key] = f[e];
    }
    __syncthreads();

    // ---- scores S(16x32) = Q(16x64) @ K^T, then *1/sqrt(64) ----
    v8f s[2];
#pragma unroll
    for (int sub = 0; sub < 2; ++sub) {
      int key = sub * 16 + l15;
      v8f c = zf;
#pragma unroll
      for (int f = 0; f < 2; ++f) {
        const unsigned short* p = &kbuf[key * 72 + f * 32 + kb0];
        c = wmma_bf16(qA[f], cat8(*(const u16x8*)p, *(const u16x8*)(p + 16)), c);
      }
      s[sub] = c * 0.125f;
    }

    // ---- online softmax (row reductions across 16-lane halves) ----
    float al[8];
#pragma unroll
    for (int i = 0; i < 8; ++i) {
      float cm = fmaxf(s[0][i], s[1][i]);
#pragma unroll
      for (int off = 8; off >= 1; off >>= 1)
        cm = fmaxf(cm, __shfl_xor(cm, off, 16));
      float nm = fmaxf(runM[i], cm);
      al[i] = __expf(runM[i] - nm);
      runM[i] = nm;
      float p0 = __expf(s[0][i] - nm);
      float p1 = __expf(s[1][i] - nm);
      s[0][i] = p0; s[1][i] = p1;
      float rs = p0 + p1;
#pragma unroll
      for (int off = 8; off >= 1; off >>= 1)
        rs += __shfl_xor(rs, off, 16);
      runL[i] = runL[i] * al[i] + rs;
    }
#pragma unroll
    for (int f = 0; f < 4; ++f)
#pragma unroll
      for (int i = 0; i < 8; ++i)
        accO[f][i] *= al[i];

    // ---- P: C-layout -> A-layout via per-wave LDS scratch ----
    unsigned short* pb = &pbuf[w * 16 * 40];
#pragma unroll
    for (int i = 0; i < 8; ++i) {
      int m = i + hi * 8;
      pb[m * 40 + l15]      = f2bf(s[0][i]);
      pb[m * 40 + 16 + l15] = f2bf(s[1][i]);
    }
    asm volatile("s_wait_dscnt 0x0" ::: "memory");   // same-wave LDS RAW fence
    u16x16 pA = cat8(*(const u16x8*)&pb[l15 * 40 + kb0],
                     *(const u16x8*)&pb[l15 * 40 + 16 + kb0]);

    // ---- O(16x64) += P(16x32) @ V(32x64) ----
#pragma unroll
    for (int f = 0; f < 4; ++f) {
      const unsigned short* p = &vbT[(f * 16 + l15) * 40 + kb0];
      accO[f] = wmma_bf16(pA, cat8(*(const u16x8*)p, *(const u16x8*)(p + 16)), accO[f]);
    }
  }

  // ---- write O / runL as bf16 ----
#pragma unroll
  for (int f = 0; f < 4; ++f) {
    int col = f * 16 + l15;
#pragma unroll
    for (int i = 0; i < 8; ++i) {
      int qrow = q0 + i + hi * 8;
      O[headoff + (size_t)qrow * D + col] = f2bf(accO[f][i] / runL[i]);
    }
  }
}

// =====================================================================
// LayerNorm over last dim (D=1024); optional bf16 secondary output.
// =====================================================================
__global__ __launch_bounds__(256)
void layernorm_k(const float* __restrict__ X, const float* __restrict__ g,
                 const float* __restrict__ bta, float* __restrict__ Y,
                 unsigned short* __restrict__ Ybf, int D)
{
  __shared__ float red[16];
  const int row = blockIdx.x;
  const int t   = threadIdx.x;
  const float* xr = X + (size_t)row * D;
  f32x4 xv = *(const f32x4*)(xr + t * 4);
  float s  = xv[0] + xv[1] + xv[2] + xv[3];
  float s2 = xv[0] * xv[0] + xv[1] * xv[1] + xv[2] * xv[2] + xv[3] * xv[3];
#pragma unroll
  for (int off = 16; off >= 1; off >>= 1) {
    s  += __shfl_xor(s,  off, 32);
    s2 += __shfl_xor(s2, off, 32);
  }
  if ((t & 31) == 0) { red[t >> 5] = s; red[8 + (t >> 5)] = s2; }
  __syncthreads();
  float ts = 0.f, ts2 = 0.f;
#pragma unroll
  for (int i = 0; i < 8; ++i) { ts += red[i]; ts2 += red[8 + i]; }
  float mu  = ts / (float)D;
  float var = ts2 / (float)D - mu * mu;
  float inv = rsqrtf(var + 1e-5f);
  f32x4 o;
#pragma unroll
  for (int e = 0; e < 4; ++e) {
    int c = t * 4 + e;
    o[e] = (xv[e] - mu) * inv * g[c] + bta[c];
  }
  *(f32x4*)(Y + (size_t)row * D + t * 4) = o;
  if (Ybf) *(u16x4*)(Ybf + (size_t)row * D + t * 4) = f4tobf(o);
}

// =====================================================================
// fp32 -> bf16 elementwise
// =====================================================================
__global__ __launch_bounds__(256)
void tobf16_k(const float* __restrict__ X, unsigned short* __restrict__ Y)
{
  size_t i = ((size_t)blockIdx.x * 256 + threadIdx.x) * 4;
  *(u16x4*)(Y + i) = f4tobf(*(const f32x4*)(X + i));
}

// =====================================================================
// Transpose + convert: In [K,N] fp32 -> Out [N,K] bf16 (32x32 LDS tiles)
// =====================================================================
__global__ __launch_bounds__(256)
void transpose_bf16_k(const float* __restrict__ In, unsigned short* __restrict__ Out,
                      int K, int N)
{
  __shared__ float tile[32][33];
  const int n0 = blockIdx.x * 32, k0 = blockIdx.y * 32;
  const int tx = threadIdx.x & 31, ty = threadIdx.x >> 5;   // ty 0..7
#pragma unroll
  for (int j = 0; j < 32; j += 8)
    tile[ty + j][tx] = In[(size_t)(k0 + ty + j) * N + n0 + tx];
  __syncthreads();
#pragma unroll
  for (int j = 0; j < 32; j += 8)
    Out[(size_t)(n0 + ty + j) * K + k0 + tx] = f2bf(tile[tx][ty + j]);
}

// =====================================================================
// Host launcher
// =====================================================================
extern "C" void kernel_launch(void* const* d_in, const int* in_sizes, int n_in,
                              void* d_out, int out_size, void* d_ws, size_t ws_size,
                              hipStream_t stream) {
  const float* x   = (const float*)d_in[0];
  const float* Wq  = (const float*)d_in[1];
  const float* bq  = (const float*)d_in[2];
  const float* Wk  = (const float*)d_in[3];
  const float* bk  = (const float*)d_in[4];
  const float* Wv  = (const float*)d_in[5];
  const float* bv  = (const float*)d_in[6];
  const float* Wo  = (const float*)d_in[7];
  const float* bo  = (const float*)d_in[8];
  const float* g1  = (const float*)d_in[9];
  const float* b1n = (const float*)d_in[10];
  const float* g2  = (const float*)d_in[11];
  const float* b2n = (const float*)d_in[12];
  const float* W1  = (const float*)d_in[13];
  const float* bf1 = (const float*)d_in[14];
  const float* W2  = (const float*)d_in[15];
  const float* bf2 = (const float*)d_in[16];
  float* out = (float*)d_out;

  const int Bc = 2, S = 2048, D = 1024, Hh = 16, DFF = 4096;
  const int M  = Bc * S;                          // 4096
  const size_t NT = (size_t)M * D;                // 4 Mi elements

  char* wp = (char*)d_ws;
  auto alloc = [&](size_t bytes) -> void* {
    void* r = (void*)wp;
    wp += (bytes + 255) & ~(size_t)255;
    return r;
  };
  unsigned short* xbf    = (unsigned short*)alloc(NT * 2);
  unsigned short* qbf    = (unsigned short*)alloc(NT * 2);
  unsigned short* kbf    = (unsigned short*)alloc(NT * 2);
  unsigned short* vbf    = (unsigned short*)alloc(NT * 2);
  unsigned short* attnbf = (unsigned short*)alloc(NT * 2);
  unsigned short* out1bf = (unsigned short*)alloc(NT * 2);
  unsigned short* hbbf   = (unsigned short*)alloc(NT * 4 * 2);
  unsigned short* WqT    = (unsigned short*)alloc((size_t)D * D * 2);
  unsigned short* WkT    = (unsigned short*)alloc((size_t)D * D * 2);
  unsigned short* WvT    = (unsigned short*)alloc((size_t)D * D * 2);
  unsigned short* WoT    = (unsigned short*)alloc((size_t)D * D * 2);
  unsigned short* W1T    = (unsigned short*)alloc((size_t)D * DFF * 2);
  unsigned short* W2T    = (unsigned short*)alloc((size_t)DFF * D * 2);
  float* res1 = (float*)alloc(NT * 4);
  float* out1 = (float*)alloc(NT * 4);
  float* res2 = (float*)alloc(NT * 4);

  dim3 blk(256);

  // ---- one-time per call: bf16 conversions + weight transposes ----
  tobf16_k<<<dim3((unsigned)(NT / 1024)), blk, 0, stream>>>(x, xbf);
  transpose_bf16_k<<<dim3(D / 32, D / 32),    blk, 0, stream>>>(Wq, WqT, D, D);
  transpose_bf16_k<<<dim3(D / 32, D / 32),    blk, 0, stream>>>(Wk, WkT, D, D);
  transpose_bf16_k<<<dim3(D / 32, D / 32),    blk, 0, stream>>>(Wv, WvT, D, D);
  transpose_bf16_k<<<dim3(D / 32, D / 32),    blk, 0, stream>>>(Wo, WoT, D, D);
  transpose_bf16_k<<<dim3(DFF / 32, D / 32),  blk, 0, stream>>>(W1, W1T, D, DFF);
  transpose_bf16_k<<<dim3(D / 32, DFF / 32),  blk, 0, stream>>>(W2, W2T, DFF, D);

  dim3 gD (D   / BN, M / BM);                     // (8, 32)
  dim3 gF1(DFF / BN, M / BM);                     // (32, 32)

  // ---- QKV projections (bf16 out only) ----
  gemm_bf16t_wmma<<<gD, blk, 0, stream>>>(xbf, WqT, bq, nullptr, nullptr, qbf, M, D, D, 0);
  gemm_bf16t_wmma<<<gD, blk, 0, stream>>>(xbf, WkT, bk, nullptr, nullptr, kbf, M, D, D, 0);
  gemm_bf16t_wmma<<<gD, blk, 0, stream>>>(xbf, WvT, bv, nullptr, nullptr, vbf, M, D, D, 0);

  // ---- attention ----
  flash_attn_wmma<<<dim3(S / 64, Hh, Bc), dim3(128), 0, stream>>>(qbf, kbf, vbf, attnbf, S, D);

  // ---- out-proj + residual(x), LN1 ----
  gemm_bf16t_wmma<<<gD, blk, 0, stream>>>(attnbf, WoT, bo, x, res1, nullptr, M, D, D, 0);
  layernorm_k<<<dim3(M), blk, 0, stream>>>(res1, g1, b1n, out1, out1bf, D);

  // ---- FFN + residual(out1), LN2 -> out ----
  gemm_bf16t_wmma<<<gF1, blk, 0, stream>>>(out1bf, W1T, bf1, nullptr, nullptr, hbbf, M, DFF, D, 1);
  gemm_bf16t_wmma<<<gD, blk, 0, stream>>>(hbbf, W2T, bf2, out1, res2, nullptr, M, D, DFF, 0);
  layernorm_k<<<dim3(M), blk, 0, stream>>>(res2, g2, b2n, out, nullptr, D);
}
